// TinySAModule_69922067579190
// MI455X (gfx1250) — matrix operand order repaired and checked by reading, not compiled, not run on hardware
//
#include <hip/hip_runtime.h>
#include <math.h>

#define DIM     128
#define NHEAD   4
#define DH      32
#define KNN     32
#define NPTS    16384
#define MCENT   4096
#define RADIUS  0.3f
#define MT      16        // centers per attention block (WMMA M)
#define CAND_MAX 1024

typedef __attribute__((ext_vector_type(16))) _Float16 v16h;
typedef __attribute__((ext_vector_type(8)))  float    v8f;

// ---------- fp16 helpers ----------
__device__ __forceinline__ unsigned short f2h_bits(float x) {
    union { _Float16 h; unsigned short s; } u;
    u.h = (_Float16)x;
    return u.s;
}

// ---------- WMMA fragment helpers (A/B in contiguous k order k = 16h + s, C/D in ISA layout, wave32) ----------
// A (16x32 f16) from LDS: lanes 0-15 row M=lane, VGPR v -> K = 2v; lanes 16-31 add +16.
__device__ __forceinline__ v16h load_a_lds(const unsigned short* lds, int stride, int col0, int lane) {
    int hh = lane >> 4, r = lane & 15;
    union { v16h v; unsigned short u[16]; } f;
#pragma unroll
    for (int v = 0; v < 8; ++v) {
        int kb = 2 * v + hh * 16;
        f.u[2 * v]     = lds[r * stride + col0 + kb];
        f.u[2 * v + 1] = lds[r * stride + col0 + kb + 1];
    }
    return f.v;
}

// B (32x16 f16): lane half selects K half (0-15 / 16-31), col N = lane&15.
// Case 1: B[k][n] = W16[(n0+n)*ldw + k0+k]  (x @ W^T shape, K contiguous)
// 16 contiguous halfs per lane -> one aligned 32-byte vector load.
__device__ __forceinline__ v16h load_b_wT_h(const unsigned short* __restrict__ W16,
                                            int ldw, int n0, int k0, int lane) {
    int hh = lane >> 4, n = lane & 15;
    const unsigned short* src = W16 + (size_t)(n0 + n) * ldw + k0 + hh * 16;
    return *(const v16h*)src;
}

// Case 2: B[k][n] = W16[(k0+k)*ldw + n0+n]  (K strided)
__device__ __forceinline__ v16h load_b_w_h(const unsigned short* __restrict__ W16,
                                           int ldw, int k0, int n0, int lane) {
    int hh = lane >> 4, n = lane & 15;
    const unsigned short* src = W16 + (size_t)(k0 + hh * 16) * ldw + n0 + n;
    union { v16h v; unsigned short u[16]; } f;
#pragma unroll
    for (int i = 0; i < 16; ++i) f.u[i] = src[(size_t)i * ldw];
    return f.v;
}

// C/D (16x16 f32): lanes 0-15 col N=lane, VGPR v -> row M=v; lanes 16-31 -> M=8+v.
__device__ __forceinline__ void store_c_f32(float* lds, int stride, int col0, int lane, v8f c) {
    int hh = lane >> 4, n = lane & 15;
#pragma unroll
    for (int v = 0; v < 8; ++v)
        lds[(v + hh * 8) * stride + col0 + n] = c[v];
}

__device__ __forceinline__ void store_c_h(unsigned short* lds, int stride, int col0, int lane, v8f c) {
    int hh = lane >> 4, n = lane & 15;
#pragma unroll
    for (int v = 0; v < 8; ++v)
        lds[(v + hh * 8) * stride + col0 + n] = f2h_bits(c[v]);
}

__device__ __forceinline__ v8f wmma16(v16h a, v16h b, v8f c) {
    return __builtin_amdgcn_wmma_f32_16x16x32_f16(false, a, false, b, (short)0, c, false, false);
}

// =====================================================================
// Kernel 0: one-shot f32 -> f16 weight conversion into workspace
// =====================================================================
__global__ void cvt_half_kernel(const float* __restrict__ src,
                                unsigned short* __restrict__ dst, int n) {
    int i = blockIdx.x * 256 + threadIdx.x;
    if (i < n) dst[i] = f2h_bits(src[i]);
}

// =====================================================================
// Kernel 1: per-center radius candidates + deterministic top-K selection
// =====================================================================
__global__ void knn_kernel(const float* __restrict__ xyz,
                           const int* __restrict__ idx_center,
                           int* __restrict__ nbr) {
    __shared__ int   s_cnt;
    __shared__ int   s_ci[CAND_MAX];
    __shared__ float s_cd[CAND_MAX];

    const int m = blockIdx.x;
    const int t = threadIdx.x;

    if (t == 0) s_cnt = 0;
    if (t < KNN) nbr[m * KNN + t] = -1;

    const int ic = idx_center[m];
    const float cx = xyz[ic * 3 + 0], cy = xyz[ic * 3 + 1], cz = xyz[ic * 3 + 2];
    __syncthreads();

    for (int n = t; n < NPTS; n += blockDim.x) {
        float dx = xyz[n * 3 + 0] - cx;
        float dy = xyz[n * 3 + 1] - cy;
        float dz = xyz[n * 3 + 2] - cz;
        float d  = sqrtf(fmaxf(dx * dx + dy * dy + dz * dz, 1e-12f));
        if (d < RADIUS) {
            int p = atomicAdd(&s_cnt, 1);
            if (p < CAND_MAX) { s_ci[p] = n; s_cd[p] = d; }
        }
    }
    __syncthreads();

    const int C = (s_cnt < CAND_MAX) ? s_cnt : CAND_MAX;
    // deterministic selection: rank by (dist, point-index)
    for (int i = t; i < C; i += blockDim.x) {
        float di = s_cd[i];
        int   ii = s_ci[i];
        int rank = 0;
        for (int j = 0; j < C; ++j) {
            float dj = s_cd[j];
            rank += (dj < di) || (dj == di && s_ci[j] < ii);
        }
        if (rank < KNN) nbr[m * KNN + rank] = ii;
    }
}

// =====================================================================
// Kernel 2: nearest center per point (argmin, first-index tie-break)
// =====================================================================
__global__ void nearest_kernel(const float* __restrict__ xyz,
                               const int* __restrict__ idx_center,
                               int* __restrict__ nearest) {
    __shared__ float s_c[256 * 3];
    const int t = threadIdx.x;
    const int n = blockIdx.x * 256 + t;

    const float px = xyz[n * 3 + 0], py = xyz[n * 3 + 1], pz = xyz[n * 3 + 2];
    float best = 3.4e38f;
    int   bi   = 0;

    for (int m0 = 0; m0 < MCENT; m0 += 256) {
        int ic = idx_center[m0 + t];
        s_c[t * 3 + 0] = xyz[ic * 3 + 0];
        s_c[t * 3 + 1] = xyz[ic * 3 + 1];
        s_c[t * 3 + 2] = xyz[ic * 3 + 2];
        __syncthreads();
#pragma unroll 4
        for (int j = 0; j < 256; ++j) {
            float dx = px - s_c[j * 3 + 0];
            float dy = py - s_c[j * 3 + 1];
            float dz = pz - s_c[j * 3 + 2];
            float d2 = dx * dx + dy * dy + dz * dz;
            if (d2 < best) { best = d2; bi = m0 + j; }
        }
        __syncthreads();
    }
    nearest[n] = bi;
}

// =====================================================================
// Kernel 3: fused attention + FFN for a tile of 16 centers (8 waves)
// =====================================================================
struct AttnSmem {
    float sXf[MT * DIM];            // center feats f32 (residual)
    float sQK[NHEAD * MT * DIM];    // per-head folded q·Wk vectors
    float s_nf[KNN * DIM];          // neighbor feats for one center
    float sU[MT * DIM];             // pre-LN activations
    float sR[MT * DIM];             // post-LN1 residual stream
    float s_sc[NHEAD * KNN];        // attention scores
    float s_mu[MT];
    float s_rs[MT];
    int   s_nidx[KNN];
    unsigned short sX16[MT * DIM];
    unsigned short sQ16[MT * DIM];
    unsigned short sW16[MT * NHEAD * DIM];  // weighted-sum vectors, stride 512
    unsigned short sO16[MT * DIM];
    unsigned short sR16[MT * DIM];
    unsigned short sH16[MT * 4 * DIM];      // FFN hidden (relu'd), stride 512
};

__global__ void attn_ffn_kernel(const float* __restrict__ feats,
                                const int* __restrict__ idx_center,
                                const unsigned short* __restrict__ Wq16,
                                const unsigned short* __restrict__ Wk16,
                                const unsigned short* __restrict__ Wv16,
                                const unsigned short* __restrict__ Wo16,
                                const float* __restrict__ bo,
                                const float* __restrict__ g1,
                                const float* __restrict__ be1,
                                const float* __restrict__ g2,
                                const float* __restrict__ be2,
                                const unsigned short* __restrict__ W116,
                                const float* __restrict__ b1f,
                                const unsigned short* __restrict__ W216,
                                const float* __restrict__ b2f,
                                const int* __restrict__ nbr,
                                float* __restrict__ c_out) {
    extern __shared__ char smem_raw[];
    AttnSmem& S = *reinterpret_cast<AttnSmem*>(smem_raw);

    const int t    = threadIdx.x;
    const int w    = t >> 5;       // wave id, 0..7
    const int lane = t & 31;
    const int m0   = blockIdx.x * MT;

    // ---- load center features ----
    for (int e = t; e < MT * DIM; e += 256) {
        int mi = e >> 7, c = e & 127;
        int ic = idx_center[m0 + mi];
        float x = feats[(size_t)ic * DIM + c];
        S.sXf[e]  = x;
        S.sX16[e] = f2h_bits(x);
    }
    // prefetch first center's neighbor feature rows while GEMM1/2 run
    if (t < KNN) {
        int id = nbr[(size_t)m0 * KNN + t];
        if (id >= 0) __builtin_prefetch(&feats[(size_t)id * DIM], 0, 0);
    }
    __syncthreads();

    // ---- GEMM1: Q = X @ Wq^T  (16x128x128) ----
    {
        int n0 = w * 16;
        v8f acc = {0.f, 0.f, 0.f, 0.f, 0.f, 0.f, 0.f, 0.f};
#pragma unroll
        for (int ks = 0; ks < 4; ++ks) {
            v16h a = load_a_lds(S.sX16, DIM, ks * 32, lane);
            v16h b = load_b_wT_h(Wq16, DIM, n0, ks * 32, lane);
            acc = wmma16(a, b, acc);
        }
        store_c_h(S.sQ16, DIM, n0, lane, acc);
    }
    __syncthreads();

    // ---- GEMM2: per-head folded projection QK_h = Q_h @ Wk_h (16x32 -> 16x128) ----
    for (int it = 0; it < 4; ++it) {
        int tile = w + 8 * it;
        int h  = tile >> 3;
        int n0 = (tile & 7) * 16;
        v16h a = load_a_lds(S.sQ16, DIM, h * DH, lane);
        v16h b = load_b_w_h(Wk16, DIM, h * DH, n0, lane);
        v8f acc = {0.f, 0.f, 0.f, 0.f, 0.f, 0.f, 0.f, 0.f};
        acc = wmma16(a, b, acc);
        store_c_f32(S.sQK + h * MT * DIM, DIM, n0, lane, acc);
    }
    __syncthreads();

    // ---- attention per center (scores, softmax, weighted sum) ----
    const float scale = 0.17677669529663687f;  // DH^-0.5
    for (int mi = 0; mi < MT; ++mi) {
        if (t < KNN) S.s_nidx[t] = nbr[(size_t)(m0 + mi) * KNN + t];
        __syncthreads();
        for (int e = t; e < KNN * DIM; e += 256) {
            int kk = e >> 7, c = e & 127;
            int id = S.s_nidx[kk];
            S.s_nf[e] = (id >= 0) ? feats[(size_t)id * DIM + c] : 0.f;
        }
        // prefetch next center's neighbor rows to hide the random gather
        if (t < KNN && (mi + 1) < MT) {
            int id = nbr[(size_t)(m0 + mi + 1) * KNN + t];
            if (id >= 0) __builtin_prefetch(&feats[(size_t)id * DIM], 0, 0);
        }
        __syncthreads();
        if (t < NHEAD * KNN) {
            int h = t >> 5, kk = t & 31;
            const float* qk = &S.sQK[(h * MT + mi) * DIM];
            const float* nf = &S.s_nf[kk * DIM];
            float s = 0.f;
#pragma unroll 4
            for (int c = 0; c < DIM; ++c) s += qk[c] * nf[c];
            S.s_sc[t] = (S.s_nidx[kk] >= 0) ? s * scale : -1e9f;
        }
        __syncthreads();
        if (t < NHEAD) {
            float mx = -3.4e38f;
            for (int k = 0; k < KNN; ++k) mx = fmaxf(mx, S.s_sc[t * KNN + k]);
            float sum = 0.f;
            for (int k = 0; k < KNN; ++k) {
                float e = __expf(S.s_sc[t * KNN + k] - mx);
                S.s_sc[t * KNN + k] = e;
                sum += e;
            }
            float inv = 1.f / sum;
            for (int k = 0; k < KNN; ++k) S.s_sc[t * KNN + k] *= inv;
        }
        __syncthreads();
        for (int o = t; o < NHEAD * DIM; o += 256) {
            int h = o >> 7, c = o & 127;
            float acc = 0.f;
#pragma unroll 8
            for (int kk = 0; kk < KNN; ++kk) acc += S.s_sc[h * KNN + kk] * S.s_nf[kk * DIM + c];
            S.sW16[mi * (NHEAD * DIM) + h * DIM + c] = f2h_bits(acc);
        }
        __syncthreads();
    }

    // ---- GEMM3: Out_h = Wsum_h @ Wv_h^T (16x128 -> 16x32 per head) ----
    {
        int h  = w >> 1;
        int nt = w & 1;
        int n0 = h * DH + nt * 16;   // column in the 128-wide output
        v8f acc = {0.f, 0.f, 0.f, 0.f, 0.f, 0.f, 0.f, 0.f};
#pragma unroll
        for (int ks = 0; ks < 4; ++ks) {
            v16h a = load_a_lds(S.sW16, NHEAD * DIM, h * DIM + ks * 32, lane);
            v16h b = load_b_wT_h(Wv16, DIM, n0, ks * 32, lane);
            acc = wmma16(a, b, acc);
        }
        store_c_h(S.sO16, DIM, n0, lane, acc);
    }
    __syncthreads();

    // ---- GEMM4: upd = Out @ Wo^T ----
    {
        int n0 = w * 16;
        v8f acc = {0.f, 0.f, 0.f, 0.f, 0.f, 0.f, 0.f, 0.f};
#pragma unroll
        for (int ks = 0; ks < 4; ++ks) {
            v16h a = load_a_lds(S.sO16, DIM, ks * 32, lane);
            v16h b = load_b_wT_h(Wo16, DIM, n0, ks * 32, lane);
            acc = wmma16(a, b, acc);
        }
        store_c_f32(S.sU, DIM, n0, lane, acc);
    }
    __syncthreads();

    // ---- LN1 + residual ----
    if (t < MT) {
        float mean = 0.f;
        for (int c = 0; c < DIM; ++c) mean += S.sU[t * DIM + c] + bo[c];
        mean *= (1.f / DIM);
        float var = 0.f;
        for (int c = 0; c < DIM; ++c) {
            float v = S.sU[t * DIM + c] + bo[c] - mean;
            var += v * v;
        }
        var *= (1.f / DIM);
        S.s_mu[t] = mean;
        S.s_rs[t] = rsqrtf(var + 1e-5f);
    }
    __syncthreads();
    for (int e = t; e < MT * DIM; e += 256) {
        int mi = e >> 7, c = e & 127;
        float u  = S.sU[e] + bo[c];
        float xn = (u - S.s_mu[mi]) * S.s_rs[mi];
        float r  = S.sXf[e] + xn * g1[c] + be1[c];
        S.sR[e]  = r;
        S.sR16[e] = f2h_bits(r);
    }
    __syncthreads();

    // ---- GEMM5: hidden = relu(x @ W1^T + b1) (16x128 -> 16x512) ----
    for (int it = 0; it < 4; ++it) {
        int n0 = (w + 8 * it) * 16;
        v8f acc = {0.f, 0.f, 0.f, 0.f, 0.f, 0.f, 0.f, 0.f};
#pragma unroll
        for (int ks = 0; ks < 4; ++ks) {
            v16h a = load_a_lds(S.sR16, DIM, ks * 32, lane);
            v16h b = load_b_wT_h(W116, DIM, n0, ks * 32, lane);
            acc = wmma16(a, b, acc);
        }
        int hh = lane >> 4, n = lane & 15;
#pragma unroll
        for (int v = 0; v < 8; ++v) {
            float val = fmaxf(acc[v] + b1f[n0 + n], 0.f);
            S.sH16[(v + hh * 8) * (4 * DIM) + n0 + n] = f2h_bits(val);
        }
    }
    __syncthreads();

    // ---- GEMM6: y = hidden @ W2^T (16x512 -> 16x128) ----
    {
        int n0 = w * 16;
        v8f acc = {0.f, 0.f, 0.f, 0.f, 0.f, 0.f, 0.f, 0.f};
#pragma unroll
        for (int ks = 0; ks < 16; ++ks) {
            v16h a = load_a_lds(S.sH16, 4 * DIM, ks * 32, lane);
            v16h b = load_b_wT_h(W216, 4 * DIM, n0, ks * 32, lane);
            acc = wmma16(a, b, acc);
        }
        store_c_f32(S.sU, DIM, n0, lane, acc);
    }
    __syncthreads();

    // ---- LN2 + residual, write center output ----
    if (t < MT) {
        float mean = 0.f;
        for (int c = 0; c < DIM; ++c) mean += S.sU[t * DIM + c] + b2f[c];
        mean *= (1.f / DIM);
        float var = 0.f;
        for (int c = 0; c < DIM; ++c) {
            float v = S.sU[t * DIM + c] + b2f[c] - mean;
            var += v * v;
        }
        var *= (1.f / DIM);
        S.s_mu[t] = mean;
        S.s_rs[t] = rsqrtf(var + 1e-5f);
    }
    __syncthreads();
    for (int e = t; e < MT * DIM; e += 256) {
        int mi = e >> 7, c = e & 127;
        float u  = S.sU[e] + b2f[c];
        float xn = (u - S.s_mu[mi]) * S.s_rs[mi];
        c_out[(size_t)(m0 + mi) * DIM + c] = S.sR[e] + xn * g2[c] + be2[c];
    }
}

// =====================================================================
// Kernel 4: out = feats + c_out[nearest]
// =====================================================================
__global__ void scatter_out_kernel(const float* __restrict__ feats,
                                   const float* __restrict__ c_out,
                                   const int* __restrict__ nearest,
                                   float* __restrict__ out) {
    int e = blockIdx.x * 256 + threadIdx.x;
    if (e < NPTS * DIM) {
        int n = e >> 7, c = e & 127;
        out[e] = feats[e] + c_out[(size_t)nearest[n] * DIM + c];
    }
}

// =====================================================================
extern "C" void kernel_launch(void* const* d_in, const int* in_sizes, int n_in,
                              void* d_out, int out_size, void* d_ws, size_t ws_size,
                              hipStream_t stream) {
    const float* xyz   = (const float*)d_in[0];
    const float* feats = (const float*)d_in[1];
    const int*   idxc  = (const int*)d_in[2];
    const float* Wq = (const float*)d_in[3];
    const float* Wk = (const float*)d_in[4];
    const float* Wv = (const float*)d_in[5];
    const float* Wo = (const float*)d_in[6];
    const float* bo = (const float*)d_in[7];
    const float* g1 = (const float*)d_in[8];
    const float* be1 = (const float*)d_in[9];
    const float* g2 = (const float*)d_in[10];
    const float* be2 = (const float*)d_in[11];
    const float* W1 = (const float*)d_in[12];
    const float* b1f = (const float*)d_in[13];
    const float* W2 = (const float*)d_in[14];
    const float* b2f = (const float*)d_in[15];
    float* out = (float*)d_out;

    char* ws = (char*)d_ws;
    size_t off = 0;
    int*   d_nbr     = (int*)(ws + off);  off += (size_t)MCENT * KNN * 4;   // 512 KB
    int*   d_nearest = (int*)(ws + off);  off += (size_t)NPTS * 4;          // 64 KB
    float* d_cout    = (float*)(ws + off); off += (size_t)MCENT * DIM * 4;  // 2 MB
    unsigned short* Wq16 = (unsigned short*)(ws + off); off += (size_t)DIM * DIM * 2;
    unsigned short* Wk16 = (unsigned short*)(ws + off); off += (size_t)DIM * DIM * 2;
    unsigned short* Wv16 = (unsigned short*)(ws + off); off += (size_t)DIM * DIM * 2;
    unsigned short* Wo16 = (unsigned short*)(ws + off); off += (size_t)DIM * DIM * 2;
    unsigned short* W116 = (unsigned short*)(ws + off); off += (size_t)4 * DIM * DIM * 2;
    unsigned short* W216 = (unsigned short*)(ws + off); off += (size_t)4 * DIM * DIM * 2;

    // one-shot weight conversion (f32 -> f16)
    cvt_half_kernel<<<(DIM * DIM + 255) / 256, 256, 0, stream>>>(Wq, Wq16, DIM * DIM);
    cvt_half_kernel<<<(DIM * DIM + 255) / 256, 256, 0, stream>>>(Wk, Wk16, DIM * DIM);
    cvt_half_kernel<<<(DIM * DIM + 255) / 256, 256, 0, stream>>>(Wv, Wv16, DIM * DIM);
    cvt_half_kernel<<<(DIM * DIM + 255) / 256, 256, 0, stream>>>(Wo, Wo16, DIM * DIM);
    cvt_half_kernel<<<(4 * DIM * DIM + 255) / 256, 256, 0, stream>>>(W1, W116, 4 * DIM * DIM);
    cvt_half_kernel<<<(4 * DIM * DIM + 255) / 256, 256, 0, stream>>>(W2, W216, 4 * DIM * DIM);

    knn_kernel<<<MCENT, 128, 0, stream>>>(xyz, idxc, d_nbr);
    nearest_kernel<<<NPTS / 256, 256, 0, stream>>>(xyz, idxc, d_nearest);
    attn_ffn_kernel<<<MCENT / MT, 256, sizeof(AttnSmem), stream>>>(
        feats, idxc, Wq16, Wk16, Wv16, Wo16, bo, g1, be1, g2, be2,
        W116, b1f, W216, b2f, d_nbr, d_cout);
    scatter_out_kernel<<<(NPTS * DIM) / 256, 256, 0, stream>>>(feats, d_cout, d_nearest, out);
}
